// LEDDecoderAttention_3813930959111
// MI455X (gfx1250) — matrix operand order repaired and checked by reading, not compiled, not run on hardware
//
#include <hip/hip_runtime.h>
#include <hip/hip_bf16.h>

typedef __attribute__((ext_vector_type(16))) _Float16 v16h;
typedef __attribute__((ext_vector_type(8)))  _Float16 v8h;
typedef __attribute__((ext_vector_type(8)))  float    v8f;

#define EMBED  768
#define NHEADS 12
#define HDIM   64
#define BSZ    2
#define SEQ    2048
#define BT     (BSZ * SEQ)   // 4096

// LDS tile strides (halves), padded so lanes 0..15 hit distinct banks
#define KSTR   72            // 32 rows x 64 halves, 144B row stride
#define VSTR   40            // 64 rows x 32 halves,  80B row stride

// ---------------------------------------------------------------------------
// WMMA wrapper:  D(16x16 f32) = A(16x32 f16) * B(32x16 f16) + C
// ---------------------------------------------------------------------------
__device__ __forceinline__ v8f wmma_f16(v16h a, v16h b, v8f c) {
    return __builtin_amdgcn_wmma_f32_16x16x32_f16(
        /*neg_a=*/false, a, /*neg_b=*/false, b,
        /*c_mod=*/(short)0, c, /*reuse_a=*/false, /*reuse_b=*/false);
}

// ---------------------------------------------------------------------------
// Fragment loaders (layouts per CDNA5 ISA 7.12.2)
// A 16x32 f16: lane r=L&15 is row M; halves 0..7 -> K = koff+0..7,
//              halves 8..15 -> K = koff+16..23, koff = (L&16)?8:0
// B 32x16 f16: lane n=L&15 is col N; halves 0..7 -> K = koff+0..7,
//              halves 8..15 -> K = koff+16..23, koff = (L&16)?8:0   (B[k][n] == Wrow[n][k] for X@W^T GEMMs)
// ---------------------------------------------------------------------------
__device__ __forceinline__ v16h a_frag_f32(const float* __restrict__ src,
                                           int row0, int ld, int k0, int lane) {
    const int r    = lane & 15;
    const int koff = (lane & 16) ? 8 : 0;
    const float* p = src + (size_t)(row0 + r) * ld + k0 + koff;
    v16h a;
#pragma unroll
    for (int e = 0; e < 8; ++e) a[e] = (_Float16)p[e];
#pragma unroll
    for (int e = 0; e < 8; ++e) a[8 + e] = (_Float16)p[16 + e];
    return a;
}

__device__ __forceinline__ v16h b_frag_f32(const float* __restrict__ src,
                                           int n0, int ld, int k0, int lane) {
    const int n    = lane & 15;
    const int koff = (lane & 16) ? 8 : 0;
    const float* p = src + (size_t)(n0 + n) * ld + k0 + koff;
    v16h b;
#pragma unroll
    for (int e = 0; e < 8; ++e) b[e] = (_Float16)p[e];
#pragma unroll
    for (int e = 0; e < 8; ++e) b[8 + e] = (_Float16)p[16 + e];
    return b;
}

__device__ __forceinline__ v16h a_frag_f16(const _Float16* __restrict__ src,
                                           int row0, int ld, int k0, int lane) {
    const int r    = lane & 15;
    const int koff = (lane & 16) ? 8 : 0;
    const _Float16* p = src + (size_t)(row0 + r) * ld + k0 + koff;
    v8h lo = *(const v8h*)p;          // 16B load
    v8h hi = *(const v8h*)(p + 16);   // 16B load
    v16h a;
#pragma unroll
    for (int e = 0; e < 8; ++e) { a[e] = lo[e]; a[8 + e] = hi[e]; }
    return a;
}

// B fragment from an LDS tile (two 16B ds loads; rows padded by caller)
__device__ __forceinline__ v16h b_frag_lds(const _Float16* base, int row0,
                                           int rowstr, int k0, int lane) {
    const int n    = lane & 15;
    const int koff = (lane & 16) ? 8 : 0;
    const _Float16* p = base + (row0 + n) * rowstr + k0 + koff;
    v8h lo = *(const v8h*)p;
    v8h hi = *(const v8h*)(p + 16);
    v16h b;
#pragma unroll
    for (int e = 0; e < 8; ++e) { b[e] = lo[e]; b[8 + e] = hi[e]; }
    return b;
}

// ---------------------------------------------------------------------------
// Kernel 1: fused QKV projection.  One wave per 16x64 output tile (one head):
// 1 A-frag feeds 4 WMMAs per k-step.
//   q = (X@Wq^T + bq) * d^-0.5  -> Qf[b][h][t][d]  (f16)
//   k =  X@Wk^T + bk            -> Kf[b][h][t][d]  (f16)
//   v =  X@Wv^T + bv            -> Vt[b][h][d][t]  (f16, transposed)
// ---------------------------------------------------------------------------
__global__ __launch_bounds__(32) void qkv_proj_kernel(
    const float* __restrict__ X,
    const float* __restrict__ Wq, const float* __restrict__ bq,
    const float* __restrict__ Wk, const float* __restrict__ bk,
    const float* __restrict__ Wv, const float* __restrict__ bv,
    _Float16* __restrict__ Qf, _Float16* __restrict__ Kf,
    _Float16* __restrict__ Vt)
{
    const int lane  = threadIdx.x & 31;
    const int m0    = blockIdx.x * 16;           // row in [0, BT)
    const int h     = blockIdx.y;                // head == 64-col group
    const int n0    = h * HDIM;
    const int which = blockIdx.z;                // 0=q 1=k 2=v
    const float* W    = (which == 0) ? Wq : (which == 1) ? Wk : Wv;
    const float* bias = (which == 0) ? bq : (which == 1) ? bk : bv;

    v8f acc[4] = {};
    for (int k0 = 0; k0 < EMBED; k0 += 32) {
        v16h a = a_frag_f32(X, m0, EMBED, k0, lane);
#pragma unroll
        for (int j = 0; j < 4; ++j) {
            v16h b = b_frag_f32(W, n0 + j * 16, EMBED, k0, lane);
            acc[j] = wmma_f16(a, b, acc[j]);
        }
    }

    const int ncol = lane & 15;
    const int half = (lane & 16) ? 8 : 0;
    const int m    = m0 + half;
    const int b    = m >> 11;                    // / SEQ  (tile never crosses)
    const int t    = m & (SEQ - 1);

    if (which == 2) {
        // transposed store: lane holds 8 consecutive t at fixed d -> 16B store
#pragma unroll
        for (int j = 0; j < 4; ++j) {
            const int   d  = j * 16 + ncol;
            const float bn = bias[n0 + d];
            v8h ov;
#pragma unroll
            for (int r = 0; r < 8; ++r) ov[r] = (_Float16)(acc[j][r] + bn);
            *(v8h*)(Vt + (((size_t)(b * NHEADS + h) * HDIM + d) * SEQ + t)) = ov;
        }
    } else {
        _Float16* dst = (which == 0) ? Qf : Kf;
        const float scale = (which == 0) ? 0.125f : 1.0f;   // HDIM^-0.5
        _Float16* rowp = dst + ((size_t)(b * NHEADS + h) * SEQ + t) * HDIM;
#pragma unroll
        for (int j = 0; j < 4; ++j) {
            const int   d  = j * 16 + ncol;
            const float bn = bias[n0 + d];
#pragma unroll
            for (int r = 0; r < 8; ++r)
                rowp[(size_t)r * HDIM + d] = (_Float16)((acc[j][r] + bn) * scale);
        }
    }
}

// ---------------------------------------------------------------------------
// Cooperative async copy of one 32-key K/V tile into LDS.
// 128 threads x 4 global_load_async_to_lds_b128 = 8KB per step.
//   K tile: 32 rows x 128B, contiguous in Kf  -> Kbuf rows padded to 144B
//   V tile: 64 rows x 64B, row stride SEQ*2   -> Vbuf rows padded to 80B
// ---------------------------------------------------------------------------
__device__ __forceinline__ void async_load_tile(
    uint32_t kbase, uint32_t vbase,
    const _Float16* __restrict__ Kh, const _Float16* __restrict__ Vh,
    int s0, int tid)
{
#pragma unroll
    for (int k = 0; k < 2; ++k) {                 // K half: 256 x 16B
        const int off   = (tid + 128 * k) * 16;   // byte offset in K tile
        const int row   = off >> 7;
        const int inrow = off & 127;
        const uint32_t lds  = kbase + (uint32_t)(row * (KSTR * 2) + inrow);
        const uint32_t goff = (uint32_t)s0 * 128u + (uint32_t)off;
        asm volatile("global_load_async_to_lds_b128 %0, %1, %2"
                     :: "v"(lds), "v"(goff), "s"(Kh) : "memory");
    }
#pragma unroll
    for (int k = 0; k < 2; ++k) {                 // V half: 256 x 16B
        const int off   = (tid + 128 * k) * 16;   // byte offset in V tile
        const int row   = off >> 6;               // d
        const int inrow = off & 63;
        const uint32_t lds  = vbase + (uint32_t)(row * (VSTR * 2) + inrow);
        const uint32_t goff = (uint32_t)row * (SEQ * 2u)
                            + (uint32_t)s0 * 2u + (uint32_t)inrow;
        asm volatile("global_load_async_to_lds_b128 %0, %1, %2"
                     :: "v"(lds), "v"(goff), "s"(Vh) : "memory");
    }
}

// ---------------------------------------------------------------------------
// Kernel 2: flash attention.  4 waves/block, each wave owns 16 queries of one
// (b,h); the block shares double-buffered K/V tiles staged into LDS by the
// async-DMA path (ASYNCcnt), cutting L2 traffic 4x.  Per 32-key step:
// 4 QK^T WMMAs -> online softmax (__shfl_xor row reductions) -> P restaged
// through a per-wave LDS tile (C-layout -> A-layout) -> 4 PV WMMAs.
// ---------------------------------------------------------------------------
__global__ __launch_bounds__(128) void flash_attn_kernel(
    const _Float16* __restrict__ Qf, const _Float16* __restrict__ Kf,
    const _Float16* __restrict__ Vt,
    const float* __restrict__ abias, const float* __restrict__ mask,
    const float* __restrict__ beta,  const float* __restrict__ beta_bias,
    _Float16* __restrict__ attn)
{
    __shared__ _Float16 Kbuf[2][32 * KSTR];      // 9216 B
    __shared__ _Float16 Vbuf[2][64 * VSTR];      // 10240 B
    __shared__ _Float16 Plds[4][16 * 32];        // 4096 B

    const int tid  = threadIdx.x;
    const int lane = tid & 31;
    const int wave = tid >> 5;
    const int bh   = blockIdx.y;                 // 0..23
    const int b    = bh / NHEADS;
    const int h    = bh % NHEADS;
    const int q0   = (blockIdx.x * 4 + wave) * 16;

    const _Float16* Qh = Qf + (size_t)bh * SEQ * HDIM;
    const _Float16* Kh = Kf + (size_t)bh * SEQ * HDIM;
    const _Float16* Vh = Vt + (size_t)bh * HDIM * SEQ;
    const float* maskb = mask  + (size_t)b  * SEQ * SEQ;
    const float* biasb = abias + (size_t)bh * SEQ;
    const float betav  = beta[0];
    const float bbias  = beta_bias[0];

    // LDS byte addresses for the async-DMA destination operand
    const uint32_t kb[2] = { (uint32_t)(uintptr_t)&Kbuf[0][0],
                             (uint32_t)(uintptr_t)&Kbuf[1][0] };
    const uint32_t vb[2] = { (uint32_t)(uintptr_t)&Vbuf[0][0],
                             (uint32_t)(uintptr_t)&Vbuf[1][0] };

    // Q A-fragments for both 32-wide k-steps over head_dim
    v16h qA0 = a_frag_f16(Qh, q0, HDIM, 0,  lane);
    v16h qA1 = a_frag_f16(Qh, q0, HDIM, 32, lane);

    const int half = (lane & 16) ? 8 : 0;
    const int ncol = lane & 15;
    _Float16* Pw = &Plds[wave][0];

    float mrow[8], lrow[8];
    v8f o0 = {}, o1 = {}, o2 = {}, o3 = {};
#pragma unroll
    for (int r = 0; r < 8; ++r) { mrow[r] = -1e30f; lrow[r] = 0.0f; }

    const int NSTEPS = SEQ / 32;                 // 64
    async_load_tile(kb[0], vb[0], Kh, Vh, 0, tid);   // prologue: tile 0

    for (int i = 0; i < NSTEPS; ++i) {
        const int cur = i & 1;
        const int s0  = i * 32;

        if (i + 1 < NSTEPS) {                    // uniform branch
            async_load_tile(kb[1 - cur], vb[1 - cur], Kh, Vh, s0 + 32, tid);
            // 4 ops outstanding for tile i+1; in-order => tile i landed
            asm volatile("s_wait_asynccnt 0x4" ::: "memory");
        } else {
            asm volatile("s_wait_asynccnt 0x0" ::: "memory");
        }
        __syncthreads();                         // tile i visible to all waves

        const _Float16* Kc = &Kbuf[cur][0];
        const _Float16* Vc = &Vbuf[cur][0];

        // ---- scores: 16 queries x 32 keys (B frags from LDS) -------------
        v16h kB0 = b_frag_lds(Kc, 0,  KSTR, 0,  lane);
        v16h kB1 = b_frag_lds(Kc, 0,  KSTR, 32, lane);
        v16h kB2 = b_frag_lds(Kc, 16, KSTR, 0,  lane);
        v16h kB3 = b_frag_lds(Kc, 16, KSTR, 32, lane);
        v8f c0 = {}, c1 = {};
        c0 = wmma_f16(qA0, kB0, c0);
        c0 = wmma_f16(qA1, kB1, c0);
        c1 = wmma_f16(qA0, kB2, c1);
        c1 = wmma_f16(qA1, kB3, c1);

        const float add0 = betav * biasb[s0 + ncol]      + bbias;
        const float add1 = betav * biasb[s0 + 16 + ncol] + bbias;

        // ---- online softmax (rows live across 16-lane halves) ------------
#pragma unroll
        for (int r = 0; r < 8; ++r) {
            const int q = q0 + half + r;
            float s0v = c0[r] + add0 + maskb[(size_t)q * SEQ + s0 + ncol];
            float s1v = c1[r] + add1 + maskb[(size_t)q * SEQ + s0 + 16 + ncol];
            float mx = fmaxf(s0v, s1v);
#pragma unroll
            for (int off = 8; off >= 1; off >>= 1)
                mx = fmaxf(mx, __shfl_xor(mx, off, 32));
            const float mnew    = fmaxf(mrow[r], mx);
            const float rescale = __expf(mrow[r] - mnew);
            mrow[r] = mnew;
            const float p0 = __expf(s0v - mnew);
            const float p1 = __expf(s1v - mnew);
            float ps = p0 + p1;
#pragma unroll
            for (int off = 8; off >= 1; off >>= 1)
                ps += __shfl_xor(ps, off, 32);
            lrow[r] = lrow[r] * rescale + ps;
            o0[r] *= rescale; o1[r] *= rescale;
            o2[r] *= rescale; o3[r] *= rescale;
            Pw[(half + r) * 32 + ncol]      = (_Float16)p0;
            Pw[(half + r) * 32 + 16 + ncol] = (_Float16)p1;
        }

        // per-wave LDS tile: wait for stores before A-layout reload
        asm volatile("s_wait_dscnt 0" ::: "memory");

        // ---- reload P as A fragment (16x32) ------------------------------
        {
            const int pr   = lane & 15;
            const int koff = (lane & 16) ? 8 : 0;
            v8h lo = *(const v8h*)&Pw[pr * 32 + koff];
            v8h hi = *(const v8h*)&Pw[pr * 32 + koff + 16];
            v16h pA;
#pragma unroll
            for (int e = 0; e < 8; ++e) { pA[e] = lo[e]; pA[8 + e] = hi[e]; }

            // ---- P @ V : V^T rows in LDS give contiguous B fragments -----
            v16h vB0 = b_frag_lds(Vc, 0,  VSTR, 0, lane);
            v16h vB1 = b_frag_lds(Vc, 16, VSTR, 0, lane);
            v16h vB2 = b_frag_lds(Vc, 32, VSTR, 0, lane);
            v16h vB3 = b_frag_lds(Vc, 48, VSTR, 0, lane);
            o0 = wmma_f16(pA, vB0, o0);
            o1 = wmma_f16(pA, vB1, o1);
            o2 = wmma_f16(pA, vB2, o2);
            o3 = wmma_f16(pA, vB3, o3);
        }

        __syncthreads();    // all waves done with buf[cur] before its reuse
    }

    // ---- epilogue: normalize, store attn[b][t][e] as f16 -----------------
#pragma unroll
    for (int r = 0; r < 8; ++r) {
        const int   q   = q0 + half + r;
        const float inv = 1.0f / lrow[r];
        const size_t rowbase = ((size_t)(b * SEQ + q)) * EMBED + h * HDIM;
        attn[rowbase      + ncol] = (_Float16)(o0[r] * inv);
        attn[rowbase + 16 + ncol] = (_Float16)(o1[r] * inv);
        attn[rowbase + 32 + ncol] = (_Float16)(o2[r] * inv);
        attn[rowbase + 48 + ncol] = (_Float16)(o3[r] * inv);
    }
}

// ---------------------------------------------------------------------------
// Kernel 3: output projection  out = attn @ Wo^T + bo   (fp32 result)
// One wave per 16x64 tile: 1 A-frag feeds 4 WMMAs per k-step.
// ---------------------------------------------------------------------------
__global__ __launch_bounds__(32) void out_proj_kernel(
    const _Float16* __restrict__ attn,
    const float* __restrict__ Wo, const float* __restrict__ bo,
    float* __restrict__ out)
{
    const int lane = threadIdx.x & 31;
    const int m0   = blockIdx.x * 16;
    const int n0   = blockIdx.y * 64;

    v8f acc[4] = {};
    for (int k0 = 0; k0 < EMBED; k0 += 32) {
        v16h a = a_frag_f16(attn, m0, EMBED, k0, lane);
#pragma unroll
        for (int j = 0; j < 4; ++j) {
            v16h b = b_frag_f32(Wo, n0 + j * 16, EMBED, k0, lane);
            acc[j] = wmma_f16(a, b, acc[j]);
        }
    }

    const int ncol = lane & 15;
    const int half = (lane & 16) ? 8 : 0;
#pragma unroll
    for (int j = 0; j < 4; ++j) {
        const int   n  = n0 + j * 16 + ncol;
        const float bn = bo[n];
#pragma unroll
        for (int r = 0; r < 8; ++r)
            out[(size_t)(m0 + half + r) * EMBED + n] = acc[j][r] + bn;
    }
}

// ---------------------------------------------------------------------------
extern "C" void kernel_launch(void* const* d_in, const int* in_sizes, int n_in,
                              void* d_out, int out_size, void* d_ws, size_t ws_size,
                              hipStream_t stream) {
    const float* X     = (const float*)d_in[0];   // [2,2048,768]
    const float* mask  = (const float*)d_in[1];   // [2,1,2048,2048]
    const float* abias = (const float*)d_in[2];   // [24,2048]
    const float* Wq    = (const float*)d_in[3];
    const float* bq    = (const float*)d_in[4];
    const float* Wk    = (const float*)d_in[5];
    const float* bk    = (const float*)d_in[6];
    const float* Wv    = (const float*)d_in[7];
    const float* bv    = (const float*)d_in[8];
    const float* Wo    = (const float*)d_in[9];
    const float* bo    = (const float*)d_in[10];
    const float* beta  = (const float*)d_in[11];
    const float* bbias = (const float*)d_in[12];
    float* out = (float*)d_out;

    // workspace: 4 f16 buffers of BT*EMBED = 3,145,728 halves each (~25 MB)
    const size_t NELEM = (size_t)BT * EMBED;
    _Float16* Qf   = (_Float16*)d_ws;
    _Float16* Kf   = Qf + NELEM;
    _Float16* Vt   = Kf + NELEM;
    _Float16* attn = Vt + NELEM;

    dim3 gridP(BT / 16, NHEADS, 3);               // 256 x 12 x 3 (16x64 tiles)
    qkv_proj_kernel<<<gridP, 32, 0, stream>>>(X, Wq, bq, Wk, bk, Wv, bv,
                                              Qf, Kf, Vt);

    dim3 gridA(SEQ / 64, BSZ * NHEADS);           // 32 x 24, 4 waves/block
    flash_attn_kernel<<<gridA, 128, 0, stream>>>(Qf, Kf, Vt, abias, mask,
                                                 beta, bbias, attn);

    dim3 gridO(BT / 16, EMBED / 64);              // 256 x 12 (16x64 tiles)
    out_proj_kernel<<<gridO, 32, 0, stream>>>(attn, Wo, bo, out);
}
